// LightweightGATLayer_58299886075963
// MI455X (gfx1250) — compile-verified, matrix operand order checked by reading, not run on hardware
//
#include <hip/hip_runtime.h>

#define N_NODES 8192
#define IN_F    256
#define OUT_F   128
#define LN_EPS  1e-5f

typedef __attribute__((ext_vector_type(2)))  float        v2f;
typedef __attribute__((ext_vector_type(8)))  float        v8f;
typedef __attribute__((ext_vector_type(16))) __bf16       v16bf;
typedef __attribute__((ext_vector_type(4)))  unsigned int v4u;
typedef __attribute__((ext_vector_type(8)))  int          v8i;
typedef __attribute__((ext_vector_type(4)))  int          v4i;

// ---------------------------------------------------------------------------
// TDM: issue a 2D tensor tile load (16 rows x 128 int32, row stride 8192 el.)
// from global memory into LDS at byte offset ldsOff.  D# layout per CDNA5 ISA
// 8.3-8.6 (group0: count/lds/global/type, group1: dims+tile+stride).
// clang-23 toolchain exposes the 6-arg builtin:
//   (v4u g0, v8i g1, v4i g2, v4i g3, v8i extra, i32 cpol)
// Must be executed by one wave only; tracked with TENSORcnt.
// ---------------------------------------------------------------------------
__device__ __forceinline__ void tdm_load_adj_tile(const int* gptr, unsigned ldsOff)
{
    unsigned long long ga = (unsigned long long)gptr;
    v4u g0 = { 1u,                                   // count=1 (valid user D#)
               ldsOff,                               // lds_addr (bytes)
               (unsigned)ga,                         // global_addr[31:0]
               (unsigned)((ga >> 32) & 0x01FFFFFFull) | (2u << 30) }; // [56:32] | type=2
    v8i g1 = { (int)0x00020000,                      // data_size=2 (4B)
               (int)((N_NODES & 0xFFFF) << 16),      // tensor_dim0 lo16
               (int)(((N_NODES >> 16) & 0xFFFF) | ((N_NODES & 0xFFFF) << 16)), // dim0 hi | dim1 lo
               (int)(((N_NODES >> 16) & 0xFFFF) | (128 << 16)), // dim1 hi | tile_dim0=128
               16,                                   // tile_dim1=16, tile_dim2=0
               N_NODES,                              // tensor_dim0_stride lo32
               0, 0 };                               // stride hi, dim1_stride=0
    v4i g2 = { 0, 0, 0, 0 };                         // 2D: dims 2/3 unused
    v4i g3 = { 0, 0, 0, 0 };
    v8i g4 = { 0, 0, 0, 0, 0, 0, 0, 0 };             // unused extra group
    __builtin_amdgcn_tensor_load_to_lds(g0, g1, g2, g3, g4, 0);
}

// ---------------------------------------------------------------------------
// Kernel 1: Wh = h @ W^T  (f32 WMMA 16x16x4), WhT (bf16, transposed), s = Wh@a
// ---------------------------------------------------------------------------
__global__ __launch_bounds__(256)
void gat_k1_wh(const float* __restrict__ h, const float* __restrict__ W,
               const float* __restrict__ a,
               __bf16* __restrict__ WhT, float* __restrict__ s)
{
    __shared__ float whTile[16 * OUT_F];
    __shared__ float sred[256];

    const int tid  = threadIdx.x;
    const int wave = tid >> 5;
    const int lane = tid & 31;
    const int half = lane >> 4;
    const int lm   = lane & 15;
    const int i0   = blockIdx.x * 16;
    const int n0   = wave * 16;

    const float* hRow = h + (size_t)(i0 + lm) * IN_F + half * 2;
    const float* wRow = W + (size_t)(n0 + lm) * IN_F + half * 2;

    v8f c = {};
    for (int k0 = 0; k0 < IN_F; k0 += 4) {
        v2f aF = *(const v2f*)(hRow + k0);
        v2f bF = *(const v2f*)(wRow + k0);
        c = __builtin_amdgcn_wmma_f32_16x16x4_f32(false, aF, false, bF,
                                                  (short)0, c, false, false);
    }
#pragma unroll
    for (int v = 0; v < 8; ++v) {
        int row = v + half * 8;
        int col = n0 + lm;
        float val = c[v];
        WhT[(size_t)col * N_NODES + (i0 + row)] = (__bf16)val;
        whTile[row * OUT_F + col] = val;
    }
    __syncthreads();
    {
        int r  = tid >> 4;
        int c0 = (tid & 15) * 8;
        float p = 0.f;
#pragma unroll
        for (int k = 0; k < 8; ++k) p += whTile[r * OUT_F + c0 + k] * a[c0 + k];
        sred[tid] = p;
    }
    __syncthreads();
    if (tid < 16) {
        float tot = 0.f;
#pragma unroll
        for (int k = 0; k < 16; ++k) tot += sred[tid * 16 + k];
        s[i0 + tid] = tot;
    }
}

// ---------------------------------------------------------------------------
// Kernel 2: per-row online softmax stats over the adjacency mask.
// ---------------------------------------------------------------------------
__global__ __launch_bounds__(256)
void gat_k2_stats(const int* __restrict__ adj, const float* __restrict__ s,
                  float* __restrict__ mrow, float* __restrict__ invz)
{
    __shared__ float mred[256];
    __shared__ float zred[256];
    const int tid = threadIdx.x;
    const int row = blockIdx.x;
    const float si = s[row];
    const int* arow = adj + (size_t)row * N_NODES;

    float m = -INFINITY, z = 0.f;
    for (int j = tid; j < N_NODES; j += 256) {
        if (j + 256 < N_NODES) __builtin_prefetch(arow + j + 256, 0, 0);
        if (arow[j] > 0) {
            float x = si + s[j];
            x = x > 0.f ? x : 0.2f * x;
            if (x > m) { z = z * __expf(m - x) + 1.f; m = x; }
            else       { z += __expf(x - m); }
        }
    }
    mred[tid] = m; zred[tid] = z;
    __syncthreads();
    for (int off = 128; off > 0; off >>= 1) {
        if (tid < off) {
            float m2 = mred[tid + off], z2 = zred[tid + off];
            float M = fmaxf(m, m2);
            if (M == -INFINITY) { z = 0.f; }
            else { z = z * __expf(m - M) + z2 * __expf(m2 - M); }
            m = M;
            mred[tid] = m; zred[tid] = z;
        }
        __syncthreads();
    }
    if (tid == 0) {
        mrow[row] = (m == -INFINITY) ? 0.f : m;
        invz[row] = (z > 0.f) ? (1.f / z) : 0.f;
    }
}

// ---------------------------------------------------------------------------
// Kernel 3: fused attention aggregate + 1/Z + LayerNorm.
// Block = 256 threads = 8 waves; block owns rows i0..i0+15 x all 128 cols.
// K loop in steps of 128 with TDM double-buffered adj tiles in LDS:
//   wave0: s_wait_tensorcnt -> barrier -> issue TDM for next tile
//   all:   build 16x128 bf16 P tile (exp from LDS adj) -> barrier
//   waves: 4x (A frag 2x ds_load_b128, B frag 2x global_load_b128, wmma bf16)
// ---------------------------------------------------------------------------
__global__ __launch_bounds__(256)
void gat_k3_aggregate(const int* __restrict__ adj,
                      const float* __restrict__ s,
                      const float* __restrict__ mrow,
                      const float* __restrict__ invz,
                      const __bf16* __restrict__ WhT,
                      const float* __restrict__ gamma,
                      const float* __restrict__ beta,
                      float* __restrict__ out)
{
    __shared__ __align__(16) int    adjTile[2][16 * 128];  // TDM dest, 2x8KB
    __shared__ __align__(16) __bf16 pA[16 * 128];          // 4KB
    __shared__ float hp[16 * OUT_F];
    __shared__ float red1[256], red2[256];
    __shared__ float sMu[16], sRstd[16], sInvZ[16];

    const int tid  = threadIdx.x;
    const int wave = tid >> 5;
    const int lane = tid & 31;
    const int half = lane >> 4;
    const int lm   = lane & 15;
    const int i0   = blockIdx.x * 16;

    // p-fill ownership: thread -> row r, 8 cols starting at c0
    const int r  = tid >> 4;
    const int c0 = (tid & 15) * 8;
    const float sA = s[i0 + r];
    const float mA = mrow[i0 + r];

    if (tid < 16) sInvZ[tid] = invz[i0 + tid];

    const int oCol = wave * 16 + lm;
    const __bf16* bBase = WhT + (size_t)oCol * N_NODES;
    const int kOff = half * 8;

    const int* adjBlk = adj + (size_t)i0 * N_NODES;
    const unsigned ldsBuf0 = (unsigned)(unsigned long long)(&adjTile[0][0]);
    const unsigned ldsBuf1 = (unsigned)(unsigned long long)(&adjTile[1][0]);

    if (wave == 0) tdm_load_adj_tile(adjBlk, ldsBuf0);

    v8f c = {};
    int it = 0;
    for (int k0 = 0; k0 < N_NODES; k0 += 128, it ^= 1) {
        if (wave == 0) __builtin_amdgcn_s_wait_tensorcnt(0);
        __syncthreads();                    // tile `it` visible; pA reusable
        if (wave == 0 && (k0 + 128) < N_NODES)
            tdm_load_adj_tile(adjBlk + k0 + 128, it ? ldsBuf0 : ldsBuf1);

        {   // P tile: p = adj ? exp(lrelu(s_i+s_j) - m_i) : 0   (bf16)
            const int* at = &adjTile[it][r * 128 + c0];
#pragma unroll
            for (int q = 0; q < 8; ++q) {
                float x = sA + s[k0 + c0 + q];
                x = x > 0.f ? x : 0.2f * x;
                pA[r * 128 + c0 + q] = (__bf16)((at[q] > 0) ? __expf(x - mA) : 0.f);
            }
        }
        __syncthreads();                    // pA ready

#pragma unroll
        for (int t = 0; t < 4; ++t) {
            union { v16bf v; v4u q[2]; } aF, bF;
            const __bf16* aPtr = pA + lm * 128 + t * 32 + kOff;
            aF.q[0] = *(const v4u*)(aPtr);        // V0-3: K = kOff..kOff+7
            aF.q[1] = *(const v4u*)(aPtr + 16);   // V4-7: K = 16+kOff..
            const __bf16* bPtr = bBase + k0 + t * 32 + kOff;
            bF.q[0] = *(const v4u*)(bPtr);
            bF.q[1] = *(const v4u*)(bPtr + 16);
            c = __builtin_amdgcn_wmma_f32_16x16x32_bf16(false, aF.v, false, bF.v,
                                                        (short)0, c, false, false);
        }
    }

    // scale by 1/Z, stage rows in LDS for LayerNorm
#pragma unroll
    for (int v = 0; v < 8; ++v) {
        int row = v + half * 8;
        hp[row * OUT_F + oCol] = c[v] * sInvZ[row];
    }
    __syncthreads();

    {
        float sum = 0.f, sq = 0.f;
#pragma unroll
        for (int k = 0; k < 8; ++k) {
            float x = hp[r * OUT_F + c0 + k];
            sum += x; sq += x * x;
        }
        red1[tid] = sum; red2[tid] = sq;
    }
    __syncthreads();
    if (tid < 16) {
        float sum = 0.f, sq = 0.f;
#pragma unroll
        for (int k = 0; k < 16; ++k) { sum += red1[tid * 16 + k]; sq += red2[tid * 16 + k]; }
        float mu  = sum * (1.f / OUT_F);
        float var = sq * (1.f / OUT_F) - mu * mu;
        sMu[tid]   = mu;
        sRstd[tid] = rsqrtf(var + LN_EPS);
    }
    __syncthreads();
    {
        float mu = sMu[r], rs = sRstd[r];
#pragma unroll
        for (int k = 0; k < 8; ++k) {
            int cI = c0 + k;
            float x = hp[r * OUT_F + cI];
            out[(size_t)(i0 + r) * OUT_F + cI] = gamma[cI] * (x - mu) * rs + beta[cI];
        }
    }
}

// ---------------------------------------------------------------------------
extern "C" void kernel_launch(void* const* d_in, const int* in_sizes, int n_in,
                              void* d_out, int out_size, void* d_ws, size_t ws_size,
                              hipStream_t stream)
{
    const float* h     = (const float*)d_in[0];
    const int*   adj   = (const int*)d_in[1];
    const float* W     = (const float*)d_in[2];
    const float* a     = (const float*)d_in[3];
    const float* gamma = (const float*)d_in[4];
    const float* beta  = (const float*)d_in[5];
    float* out = (float*)d_out;

    char*   ws   = (char*)d_ws;
    __bf16* WhT  = (__bf16*)ws;                                      // 2 MB
    float*  s    = (float*)(ws + (size_t)OUT_F * N_NODES * sizeof(__bf16));
    float*  mrow = s + N_NODES;
    float*  invz = mrow + N_NODES;

    gat_k1_wh       <<<N_NODES / 16, 256, 0, stream>>>(h, W, a, WhT, s);
    gat_k2_stats    <<<N_NODES,      256, 0, stream>>>(adj, s, mrow, invz);
    gat_k3_aggregate<<<N_NODES / 16, 256, 0, stream>>>(adj, s, mrow, invz, WhT,
                                                       gamma, beta, out);
}
